// DimeNetInteraction_39393440039342
// MI455X (gfx1250) — matrix operand order repaired and checked by reading, not compiled, not run on hardware
//
#include <hip/hip_runtime.h>
#include <hip/hip_bf16.h>

// ---------------------------------------------------------------------------
// DimeNet-style interaction block, fused for gfx1250 (CDNA5, wave32, WMMA).
// B=4, A=64, N=64, H=128, S=7.  One workgroup (8 waves) per (b, j) pair.
// Dominant GEMM (1M x 128 x 128) runs on v_wmma_f32_16x16x32_f16 with f32
// accumulation; the huge (B,A,N,N,H) intermediate never leaves registers/LDS.
// silu uses the HW approximate reciprocal (v_rcp_f32).  The angle-hidden
// staging tile is double-buffered (one barrier per tile), and the masked
// edge-feature values for the k-contraction are preloaded into registers
// (they are i-invariant), so the hot loop touches LDS only for the WMMA
// A-fragments and the staging-tile stores.
// ---------------------------------------------------------------------------

typedef _Float16 v8h  __attribute__((ext_vector_type(8)));
typedef _Float16 v16h __attribute__((ext_vector_type(16)));
typedef float    v8f  __attribute__((ext_vector_type(8)));

#define LSTRIDE 136   // padded row stride (halves) for LDS f16 matrices
#define NFRAG_AW1 0   // fragment table offsets (fragments of 512 halves)
#define NFRAG_AW2 8
#define NFRAG_MW1 40
#define NFRAG_MW2 104
#define NFRAG_TOT 136

__device__ __forceinline__ float silu_f(float x) {
    // x * sigmoid(x); v_exp_f32 + v_rcp_f32 (both TRANS, co-execute with VALU)
    float e = __expf(-x);
    return x * __builtin_amdgcn_rcpf(1.0f + e);
}

__device__ __forceinline__ v16h load_frag_g(const _Float16* pack, int frag, int lane) {
    const _Float16* p = pack + ((size_t)frag << 9) + (lane << 4);
    v8h lo = *(const v8h*)p;
    v8h hi = *(const v8h*)(p + 8);
    v16h r;
#pragma unroll
    for (int e = 0; e < 8; ++e) { r[e] = lo[e]; r[e + 8] = hi[e]; }
    return r;
}

__device__ __forceinline__ v16h load_frag_lds(const _Float16* base) {
    // base points at row*LSTRIDE + 32*kstep + (hi?8:0); A-layout: K{+0..7,+16..23}
    v8h lo = *(const v8h*)base;
    v8h hi = *(const v8h*)(base + 16);
    v16h r;
#pragma unroll
    for (int e = 0; e < 8; ++e) { r[e] = lo[e]; r[e + 8] = hi[e]; }
    return r;
}

// ---------------------------------------------------------------------------
// Prep: convert f32 weights to f16 fragments in WMMA B-operand layout.
// Fragment (kstep s, col-tile t): lane l holds 16 halves = W[32s + (kk<8 ? 8h+kk : 16+8h+kk-8)][16t + (l&15)], h = l>>4
// ---------------------------------------------------------------------------
__global__ void dimenet_pack_weights(const float* __restrict__ aw1,
                                     const float* __restrict__ aw2,
                                     const float* __restrict__ mw1,
                                     const float* __restrict__ mw2,
                                     _Float16* __restrict__ pack) {
    int gid = blockIdx.x * blockDim.x + threadIdx.x;
    if (gid >= NFRAG_TOT * 512) return;
    int f    = gid >> 9;
    int r    = gid & 511;
    int lane = r >> 4;
    int kk   = r & 15;
    const float* W; int Kdim, s, t;
    if (f < NFRAG_AW2)      { W = aw1; Kdim = 7;   s = 0;              t = f; }
    else if (f < NFRAG_MW1) { int g = f - NFRAG_AW2; W = aw2; Kdim = 128; s = g >> 3; t = g & 7; }
    else if (f < NFRAG_MW2) { int g = f - NFRAG_MW1; W = mw1; Kdim = 256; s = g >> 3; t = g & 7; }
    else                    { int g = f - NFRAG_MW2; W = mw2; Kdim = 128; s = g >> 3; t = g & 7; }
    int K = 32 * s + ((kk < 8) ? ((lane >> 4) * 8 + kk) : (16 + (lane >> 4) * 8 + kk - 8));
    int N = 16 * t + (lane & 15);
    float v = (K < Kdim) ? W[K * 128 + N] : 0.0f;
    pack[gid] = (_Float16)v;
}

// ---------------------------------------------------------------------------
// Main fused kernel: one block per (b, j).
// ---------------------------------------------------------------------------
__global__ __launch_bounds__(256)
void dimenet_interaction_kernel(const float* __restrict__ EF,           // (B,A,N,H) f32
                                const float* __restrict__ DIR,          // (B,A,N,3) f32
                                const unsigned char* __restrict__ EM,   // (B,A,N) bool
                                const float* __restrict__ ab1,
                                const float* __restrict__ ab2,
                                const float* __restrict__ mb1,
                                const float* __restrict__ mb2,
                                const _Float16* __restrict__ pack,
                                float* __restrict__ out) {
    __shared__ _Float16 sEF[64 * LSTRIDE];       // edge_features[b,j] f16
    __shared__ _Float16 sNM[64 * LSTRIDE];       // neighbour_messages f16
    __shared__ _Float16 sY [64 * LSTRIDE];       // MLP hidden f16
    __shared__ _Float16 sH [2][16 * LSTRIDE];    // double-buffered angle-hidden staging
    __shared__ float    sDir[64 * 4];
    __shared__ float    sMask[64];
    __shared__ float    sB1[128], sB2[128], sB3[128], sB4[128];

    const int bj   = blockIdx.x;             // b*64 + j
    const int tid  = threadIdx.x;
    const int lane = tid & 31;
    const int w    = tid >> 5;               // wave id 0..7 -> column tile
    const int cl   = lane & 15;              // column within tile / A-row
    const int hi   = lane >> 4;              // lane half (0/1)
    const int hb   = w * 16;                 // h column base

    const float* ef  = EF  + (size_t)bj * 64 * 128;
    const float* dir = DIR + (size_t)bj * 64 * 3;
    const unsigned char* em = EM + (size_t)bj * 64;

    // ---- stage per-(b,j) inputs ----
    for (int idx = tid; idx < 64 * 128; idx += 256) {
        int n = idx >> 7, h = idx & 127;
        sEF[n * LSTRIDE + h] = (_Float16)ef[idx];
    }
    for (int idx = tid; idx < 64; idx += 256) {
        sDir[idx * 4 + 0] = dir[idx * 3 + 0];
        sDir[idx * 4 + 1] = dir[idx * 3 + 1];
        sDir[idx * 4 + 2] = dir[idx * 3 + 2];
        sMask[idx] = em[idx] ? 1.0f : 0.0f;
    }
    for (int idx = tid; idx < 128; idx += 256) {
        sB1[idx] = ab1[idx]; sB2[idx] = ab2[idx];
        sB3[idx] = mb1[idx]; sB4[idx] = mb2[idx];
    }
    __syncthreads();

    // ---- per-wave resident B fragments: aw1 tile w, aw2 tile w (4 k-steps) ----
    const v16h B1 = load_frag_g(pack, NFRAG_AW1 + w, lane);
    v16h B2[4];
#pragma unroll
    for (int s = 0; s < 4; ++s) B2[s] = load_frag_g(pack, NFRAG_AW2 + s * 8 + w, lane);

    const float cb1 = sB1[hb + cl];
    const float cb2 = sB2[hb + cl];

    // ---- preload mask-scaled edge features for the k-contraction ----
    // efv[kt][p] = edge_mask[k] * EF[k][hb+cl], k = 16*kt + p + 8*hi  (i-invariant)
    float efv[4][8];
#pragma unroll
    for (int kt = 0; kt < 4; ++kt)
#pragma unroll
        for (int p = 0; p < 8; ++p) {
            const int k = kt * 16 + p + hi * 8;
            efv[kt][p] = sMask[k] * (float)sEF[k * LSTRIDE + hb + cl];
        }

    // ---- Phase 1: angle embedding + k-contraction ----
    // One barrier per tile: write sH[buf]; barrier; read sH[buf]; next tile flips buf.
    // A wave signals barrier n+1 only after its tile-n reads fed that tile's WMMAs,
    // so the tile-n+2 writer of the same buffer can never race them.
    for (int i = 0; i < 64; ++i) {
        const float dix = sDir[i * 4 + 0];
        const float diy = sDir[i * 4 + 1];
        const float diz = sDir[i * 4 + 2];
        const float mi  = sMask[i];
        float nmacc = 0.0f;

#pragma unroll
        for (int kt = 0; kt < 4; ++kt) {
            const int kb = kt * 16;
            _Float16* Hbuf = sH[kt & 1];

            // A1: Chebyshev basis T_1..T_7(cos angle) for rows k = kb..kb+15
            v16h A1;
#pragma unroll
            for (int e = 0; e < 16; ++e) A1[e] = (_Float16)0.0f;
            if (lane < 16) {
                const int k = kb + lane;
                float x = dix * sDir[k * 4 + 0]
                        + diy * sDir[k * 4 + 1]
                        + diz * sDir[k * 4 + 2];
                x = fminf(fmaxf(x, -1.0f + 1e-7f), 1.0f - 1e-7f);
                float tp = 1.0f, tc = x;
                A1[0] = (_Float16)x;
#pragma unroll
                for (int s = 1; s < 7; ++s) {
                    float tn = 2.0f * x * tc - tp;
                    A1[s] = (_Float16)tn;
                    tp = tc; tc = tn;
                }
            }
            v8f c1;
#pragma unroll
            for (int e = 0; e < 8; ++e) c1[e] = cb1;
            v8f d1 = __builtin_amdgcn_wmma_f32_16x16x32_f16(false, A1, false, B1,
                                                            (short)0, c1, false, false);
            // silu -> f16 -> staging tile (this wave's 16 columns)
#pragma unroll
            for (int p = 0; p < 8; ++p) {
                float v = silu_f(d1[p]);
                Hbuf[(p + hi * 8) * LSTRIDE + hb + cl] = (_Float16)v;
            }
            __syncthreads();   // full h tile (16 x 128) visible

            // A2 fragments for the 4 K-steps of h @ aw2
            v16h A2[4];
#pragma unroll
            for (int s = 0; s < 4; ++s)
                A2[s] = load_frag_lds(Hbuf + cl * LSTRIDE + 32 * s + hi * 8);

            v8f acc;
#pragma unroll
            for (int e = 0; e < 8; ++e) acc[e] = cb2;
#pragma unroll
            for (int s = 0; s < 4; ++s)
                acc = __builtin_amdgcn_wmma_f32_16x16x32_f16(false, A2[s], false, B2[s],
                                                             (short)0, acc, false, false);

            // silu + diagonal exclusion + reduce over k (tile rows)
            float part = 0.0f;
#pragma unroll
            for (int p = 0; p < 8; ++p) {
                const int k = kb + p + hi * 8;
                float v = silu_f(acc[p]);
                float e = (k == i) ? 0.0f : efv[kt][p];
                part = fmaf(v, e, part);
            }
            part += __shfl_xor(part, 16, 32);
            nmacc += part;
        }
        if (lane < 16) sNM[i * LSTRIDE + hb + cl] = (_Float16)(nmacc * mi);
    }
    __syncthreads();

    // ---- Phase 2: cat(EF, NM) @ mw1 -> silu ----
    // mw1 fragments for this wave's column tile, resident in registers
    v16h Bm1[8];
#pragma unroll
    for (int s = 0; s < 8; ++s) Bm1[s] = load_frag_g(pack, NFRAG_MW1 + s * 8 + w, lane);

    const float cbm1 = sB3[hb + cl];
    for (int rt2 = 0; rt2 < 4; ++rt2) {
        const int rb = rt2 * 16;
        v8f acc;
#pragma unroll
        for (int e = 0; e < 8; ++e) acc[e] = cbm1;
#pragma unroll
        for (int s = 0; s < 8; ++s) {
            const _Float16* src = (s < 4) ? sEF : sNM;
            const int ks = s & 3;
            v16h Af = load_frag_lds(src + (rb + cl) * LSTRIDE + 32 * ks + hi * 8);
            acc = __builtin_amdgcn_wmma_f32_16x16x32_f16(false, Af, false, Bm1[s],
                                                         (short)0, acc, false, false);
        }
#pragma unroll
        for (int p = 0; p < 8; ++p) {
            float v = silu_f(acc[p]);
            sY[(rb + p + hi * 8) * LSTRIDE + hb + cl] = (_Float16)v;
        }
    }
    __syncthreads();

    // ---- Phase 3: Y @ mw2 -> silu -> residual + edge mask -> out ----
    v16h Bm2[4];
#pragma unroll
    for (int s = 0; s < 4; ++s) Bm2[s] = load_frag_g(pack, NFRAG_MW2 + s * 8 + w, lane);

    const float cbm2 = sB4[hb + cl];
    for (int rt2 = 0; rt2 < 4; ++rt2) {
        const int rb = rt2 * 16;
        v8f acc;
#pragma unroll
        for (int e = 0; e < 8; ++e) acc[e] = cbm2;
#pragma unroll
        for (int s = 0; s < 4; ++s) {
            v16h Af = load_frag_lds(sY + (rb + cl) * LSTRIDE + 32 * s + hi * 8);
            acc = __builtin_amdgcn_wmma_f32_16x16x32_f16(false, Af, false, Bm2[s],
                                                         (short)0, acc, false, false);
        }
#pragma unroll
        for (int p = 0; p < 8; ++p) {
            const int i2 = rb + p + hi * 8;
            float v = silu_f(acc[p]);
            const int off = i2 * 128 + hb + cl;
            out[(size_t)bj * 8192 + off] = ef[off] + v * sMask[i2];
        }
    }
}

// ---------------------------------------------------------------------------
// Launch
// ---------------------------------------------------------------------------
extern "C" void kernel_launch(void* const* d_in, const int* in_sizes, int n_in,
                              void* d_out, int out_size, void* d_ws, size_t ws_size,
                              hipStream_t stream) {
    (void)in_sizes; (void)n_in; (void)out_size; (void)ws_size;
    const float* edge_features = (const float*)d_in[0];
    const float* directions    = (const float*)d_in[1];
    const unsigned char* emask = (const unsigned char*)d_in[2];
    const float* aw1 = (const float*)d_in[3];
    const float* ab1 = (const float*)d_in[4];
    const float* aw2 = (const float*)d_in[5];
    const float* ab2 = (const float*)d_in[6];
    const float* mw1 = (const float*)d_in[7];
    const float* mb1 = (const float*)d_in[8];
    const float* mw2 = (const float*)d_in[9];
    const float* mb2 = (const float*)d_in[10];
    float* out = (float*)d_out;
    _Float16* pack = (_Float16*)d_ws;   // needs 136*512*2 = 139,264 bytes

    const int packElems = NFRAG_TOT * 512;
    dimenet_pack_weights<<<(packElems + 255) / 256, 256, 0, stream>>>(
        aw1, aw2, mw1, mw2, pack);

    dimenet_interaction_kernel<<<256, 256, 0, stream>>>(
        edge_features, directions, emask, ab1, ab2, mb1, mb2, pack, out);
}
